// JCAF_44392781972143
// MI455X (gfx1250) — matrix operand order repaired and checked by reading, not compiled, not run on hardware
//
#include <hip/hip_runtime.h>
#include <hip/hip_bf16.h>
#include <math.h>

// ---------------------------------------------------------------------------
// Types for CDNA5 WMMA (wave32): v_wmma_f32_16x16x32_bf16
// ---------------------------------------------------------------------------
typedef __attribute__((ext_vector_type(16))) __bf16 v16bf;
typedef __attribute__((ext_vector_type(8)))  __bf16 v8bf;
typedef __attribute__((ext_vector_type(8)))  float  v8f;

#define BT 64    // block tile M/N
#define BK 32    // K step
#define LDP 40   // padded LDS row (32 + 8 elems); 80B pitch = 5*16B -> 16B-aligned chunks

// Fragment loader: 16-bit A-matrix 16x32 layout (ISA 7.12.2).
// lanes 0-15: m=lane, V0-3 -> K=0..7, V4-7 -> K=16..23
// lanes16-31: m=lane-16, V0-3 -> K=8..15, V4-7 -> K=24..31
// = two contiguous 16B chunks at elem offsets grp*8 and 16+grp*8 -> ds_load_b128 x2
__device__ __forceinline__ v16bf frag_A(const __bf16* smem, int row0, int ld) {
  int lane = threadIdx.x & 31;
  int grp  = lane >> 4;
  int m    = lane & 15;
  const __bf16* rp = smem + (row0 + m) * ld;
  v8bf lo = *(const v8bf*)(rp + grp * 8);
  v8bf hi = *(const v8bf*)(rp + 16 + grp * 8);
  return __builtin_shufflevector(lo, hi, 0, 1, 2, 3, 4, 5, 6, 7,
                                 8, 9, 10, 11, 12, 13, 14, 15);
}

// 16-bit B-matrix 32x16 layout: lane n holds K = grp*8..grp*8+7 and 16+grp*8..16+grp*8+7
// (smem holds B transposed: Bt[n][k]) -> two 16B chunks = ds_load_b128 x2
__device__ __forceinline__ v16bf frag_B(const __bf16* smem, int row0, int ld) {
  int lane = threadIdx.x & 31;
  int grp  = lane >> 4;
  int n    = lane & 15;
  const __bf16* rp = smem + (row0 + n) * ld;
  v8bf lo = *(const v8bf*)(rp + grp * 8);
  v8bf hi = *(const v8bf*)(rp + 16 + grp * 8);
  return __builtin_shufflevector(lo, hi, 0, 1, 2, 3, 4, 5, 6, 7,
                                 8, 9, 10, 11, 12, 13, 14, 15);
}

// ---------------------------------------------------------------------------
// Generic batched bf16 x bf16 -> f32 GEMM.
// Requirements (all launches satisfy them): sAk == 1; sBk == 1 or sBn == 1;
// all leading strides multiples of 8 elements; M,N multiples of 64; K of 32.
// OUTMODE: 0 = f32 store (+optional f32 addend, act); if sOm==1 && sOn!=1 the
//              64x64 tile is transposed through LDS for coalesced stores.
//          1 = bf16 store (+optional f32 addend, act)
//          2 = bias + pairwise channel sum -> f32 [M, N/2] + global sum(x^2)
// act: 0 none, 1 tanh, 2 relu
// ---------------------------------------------------------------------------
template <int OUTMODE>
__global__ __launch_bounds__(128) void gemm_bf16_k(
    const __bf16* __restrict__ A, long long sAm, long long sAb,
    const __bf16* __restrict__ B, long long sBk, long long sBn, long long sBb,
    void* __restrict__ Out, long long sOm, long long sOn, long long sOb,
    const float* __restrict__ bias,
    const float* __restrict__ Add, long long sAdm, long long sAdn, long long sAdb,
    float* __restrict__ sumsq,
    int Kd, float alpha, int act)
{
  __shared__ __align__(16) union {
    struct { __bf16 A[BT][LDP]; __bf16 B[BT][LDP]; } st;
    float c[BT][BT + 1];                 // f32 tile for transposed epilogue
  } sm;
  __shared__ float red[128];

  const int tid  = threadIdx.x;
  const int wave = tid >> 5;
  const int wm   = wave >> 1;
  const int wn   = wave & 1;
  const int bN0  = blockIdx.x * BT;
  const int bM0  = blockIdx.y * BT;
  const long long bb = blockIdx.z;

  const __bf16* Ab = A + bb * sAb;
  const __bf16* Bb = B + bb * sBb;

  v8f acc[2][2] = {};

  for (int k0 = 0; k0 < Kd; k0 += BK) {
    __syncthreads();
    // ---- stage A (64 x 32), k-contiguous: 16B vector copies ----
    for (int idx = tid; idx < (BT * BK) / 8; idx += 128) {   // 256 chunks
      int r = idx >> 2;            // 0..63
      int c = (idx & 3) * 8;       // 0,8,16,24
      const __bf16* gp = Ab + (long long)(bM0 + r) * sAm + (k0 + c);
      *(uint4*)&sm.st.A[r][c] = *(const uint4*)gp;
      if (k0 + BK < Kd) __builtin_prefetch(gp + BK, 0, 0);
    }
    // ---- stage B transposed (Bt[n][k], 64 x 32) ----
    if (sBk == 1) {
      // k-contiguous: 16B vector copy straight into Bt rows
      for (int idx = tid; idx < (BT * BK) / 8; idx += 128) {
        int n = idx >> 2;
        int c = (idx & 3) * 8;
        const __bf16* gp = Bb + (long long)(bN0 + n) * sBn + (k0 + c);
        *(uint4*)&sm.st.B[n][c] = *(const uint4*)gp;
        if (k0 + BK < Kd) __builtin_prefetch(gp + BK, 0, 0);
      }
    } else {
      // n-contiguous: 16B vector global load, scalar scatter into Bt
      for (int idx = tid; idx < (BT * BK) / 8; idx += 128) {
        int c = idx >> 3;            // k 0..31
        int n = (idx & 7) * 8;       // n 0..56
        const __bf16* gp = Bb + (long long)(k0 + c) * sBk + (bN0 + n);
        uint4 u = *(const uint4*)gp;
        __align__(16) __bf16 t8[8];
        *(uint4*)t8 = u;
#pragma unroll
        for (int j = 0; j < 8; ++j) sm.st.B[n + j][c] = t8[j];
        if (k0 + BK < Kd) __builtin_prefetch(gp + BK * sBk, 0, 0);
      }
    }
    __syncthreads();

    v16bf a0 = frag_A(&sm.st.A[0][0], wm * 32,      LDP);
    v16bf a1 = frag_A(&sm.st.A[0][0], wm * 32 + 16, LDP);
    v16bf b0 = frag_B(&sm.st.B[0][0], wn * 32,      LDP);
    v16bf b1 = frag_B(&sm.st.B[0][0], wn * 32 + 16, LDP);

    acc[0][0] = __builtin_amdgcn_wmma_f32_16x16x32_bf16(false, a0, false, b0, (short)0, acc[0][0], false, false);
    acc[0][1] = __builtin_amdgcn_wmma_f32_16x16x32_bf16(false, a0, false, b1, (short)0, acc[0][1], false, false);
    acc[1][0] = __builtin_amdgcn_wmma_f32_16x16x32_bf16(false, a1, false, b0, (short)0, acc[1][0], false, false);
    acc[1][1] = __builtin_amdgcn_wmma_f32_16x16x32_bf16(false, a1, false, b1, (short)0, acc[1][1], false, false);
  }

  // Epilogue. C/D layout: VGPR r, lanes<16 -> M=r (grp 0) / lanes>=16 -> M=r+8; N=lane&15.
  const int lane = tid & 31;
  const int grp  = lane >> 4;
  const int nl   = lane & 15;

  if (OUTMODE == 0 && sOm == 1 && sOn != 1) {
    // Transposed store: round-trip through LDS so stores are coalesced along m.
    __syncthreads();                       // staging buffers dead -> reuse as f32 tile
#pragma unroll
    for (int mi = 0; mi < 2; ++mi)
#pragma unroll
      for (int ni = 0; ni < 2; ++ni)
#pragma unroll
        for (int r = 0; r < 8; ++r)
          sm.c[wm * 32 + mi * 16 + grp * 8 + r][wn * 32 + ni * 16 + nl] =
              acc[mi][ni][r] * alpha;
    __syncthreads();
    for (int idx = tid; idx < BT * BT; idx += 128) {
      int mm = idx & 63;                   // fast dim = m (unit stride in Out)
      int nn = idx >> 6;
      float v = sm.c[mm][nn];
      if (Add) v += Add[(long long)(bM0 + mm) * sAdm + (long long)(bN0 + nn) * sAdn + bb * sAdb];
      ((float*)Out)[(long long)(bM0 + mm) * sOm + (long long)(bN0 + nn) * sOn + bb * sOb] = v;
    }
    return;
  }

  float ss = 0.f;
#pragma unroll
  for (int mi = 0; mi < 2; ++mi) {
#pragma unroll
    for (int ni = 0; ni < 2; ++ni) {
#pragma unroll
      for (int r = 0; r < 8; ++r) {
        int m = bM0 + wm * 32 + mi * 16 + grp * 8 + r;
        int n = bN0 + wn * 32 + ni * 16 + nl;
        float v = acc[mi][ni][r] * alpha;
        if (OUTMODE == 2) {
          v += bias[n];
          ss += v * v;
          float pr = v + __shfl_xor(v, 1, 32);   // adjacent-channel pair sum
          if ((n & 1) == 0)
            ((float*)Out)[(long long)m * sOm + (long long)(n >> 1) * sOn] = pr;
        } else {
          if (Add) v += Add[(long long)m * sAdm + (long long)n * sAdn + bb * sAdb];
          if (act == 1) v = tanhf(v);
          else if (act == 2) v = fmaxf(v, 0.f);
          long long o = (long long)m * sOm + (long long)n * sOn + bb * sOb;
          if (OUTMODE == 0) ((float*)Out)[o] = v;
          else              ((__bf16*)Out)[o] = (__bf16)v;
        }
      }
    }
  }

  if (OUTMODE == 2) {
    red[tid] = ss;
    __syncthreads();
    for (int s = 64; s > 0; s >>= 1) {
      if (tid < s) red[tid] += red[tid + s];
      __syncthreads();
    }
    if (tid == 0) atomicAdd(sumsq, red[0]);
  }
}

// ---------------------------------------------------------------------------
// Elementwise helpers
// ---------------------------------------------------------------------------
__global__ void zero_k(float* p, int n) {
  int i = blockIdx.x * blockDim.x + threadIdx.x;
  if (i < n) p[i] = 0.f;
}

__global__ void cvt_k(const float* __restrict__ in, __bf16* __restrict__ out, long long n) {
  long long stride = (long long)gridDim.x * blockDim.x;
  for (long long i = (long long)blockIdx.x * blockDim.x + threadIdx.x; i < n; i += stride)
    out[i] = (__bf16)in[i];
}

// [B, L, D] f32 -> [B, D, L] bf16 transpose (32x32 LDS tiles, coalesced both sides)
__global__ void trans_k(const float* __restrict__ in, __bf16* __restrict__ out) {
  __shared__ float t[32][33];
  int b  = blockIdx.z;
  int l0 = blockIdx.y * 32;
  int d0 = blockIdx.x * 32;
  int tx = threadIdx.x, ty = threadIdx.y;   // (32, 8)
  for (int i = ty; i < 32; i += 8)
    t[i][tx] = in[((long long)b * 1024 + (l0 + i)) * 128 + (d0 + tx)];
  __syncthreads();
  for (int i = ty; i < 32; i += 8)
    out[((long long)b * 128 + (d0 + i)) * 1024 + (l0 + tx)] = (__bf16)t[tx][i];
}

__global__ void scalars_k(float* s) {
  // s[0]=sum f1^2, s[1]=sum f2^2 -> s[2]=n1/(n1+n2), s[3]=n2/(n1+n2)
  float n1 = sqrtf(s[0]);
  float n2 = sqrtf(s[1]);
  float t  = n1 + n2;
  s[2] = n1 / t;
  s[3] = n2 / t;
}

// z = w1*p1 + w2*p2 ; zz[b,d] = sum_l z^2   (grid: (64 batches, 8 l-chunks), 128 thr = d)
__global__ void zkern(const float* __restrict__ p1, const float* __restrict__ p2,
                      const float* __restrict__ scal, float* __restrict__ z,
                      float* __restrict__ zz)
{
  int b = blockIdx.x, lc = blockIdx.y, d = threadIdx.x;
  float w1 = scal[2], w2 = scal[3];
  float ss = 0.f;
  long long base = ((long long)b * 1024 + (long long)lc * 128) * 128 + d;
  for (int l = 0; l < 128; ++l) {
    long long i = base + (long long)l * 128;
    float v = w1 * p1[i] + w2 * p2[i];
    z[i] = v;
    ss += v * v;
  }
  atomicAdd(&zz[b * 128 + d], ss);
}

// G = z / max(||z||_L, 1e-12)  -> bf16
__global__ void norm_k(const float* __restrict__ z, const float* __restrict__ zz,
                       __bf16* __restrict__ G, long long n)
{
  long long stride = (long long)gridDim.x * blockDim.x;
  for (long long i = (long long)blockIdx.x * blockDim.x + threadIdx.x; i < n; i += stride) {
    int d = (int)(i & 127);
    int b = (int)(i >> 17);             // L*D = 131072 = 2^17
    float dn = fmaxf(sqrtf(zz[b * 128 + d]), 1e-12f);
    G[i] = (__bf16)(z[i] / dn);
  }
}

// G_ag = concat(aud, G) ; G_lg = concat(txt, G)  over channel dim (256)
__global__ void concat_k(const __bf16* __restrict__ txtb, const __bf16* __restrict__ audb,
                         const __bf16* __restrict__ Gb, __bf16* __restrict__ Gag,
                         __bf16* __restrict__ Glg, long long n)
{
  long long stride = (long long)gridDim.x * blockDim.x;
  for (long long i = (long long)blockIdx.x * blockDim.x + threadIdx.x; i < n; i += stride) {
    int c = (int)(i & 255);
    long long bl = i >> 8;
    if (c < 128) {
      Gag[i] = audb[bl * 128 + c];
      Glg[i] = txtb[bl * 128 + c];
    } else {
      __bf16 g = Gb[bl * 128 + (c - 128)];
      Gag[i] = g;
      Glg[i] = g;
    }
  }
}

// ---------------------------------------------------------------------------
// Host launcher
// ---------------------------------------------------------------------------
static const long long Bn = 64, Ln = 1024, Dn = 128, Kn = 256;
static const long long BLD = Bn * Ln * Dn;            // 8388608
static const long long BL2D = Bn * Ln * 2 * Dn;       // 16777216

extern "C" void kernel_launch(void* const* d_in, const int* in_sizes, int n_in,
                              void* d_out, int out_size, void* d_ws, size_t ws_size,
                              hipStream_t stream)
{
  (void)in_sizes; (void)n_in; (void)out_size; (void)ws_size;

  const float* txt = (const float*)d_in[0];
  const float* aud = (const float*)d_in[1];
  const float* vis = (const float*)d_in[2];
  const float* Wi  = (const float*)d_in[3];
  const float* bi  = (const float*)d_in[4];
  const float* Wq  = (const float*)d_in[5];
  const float* bq  = (const float*)d_in[6];
  // branch order: 0=txt, 1=aud, 2=vis  (vis uses G_ag: original bug preserved)
  const float* Waff[3] = { (const float*)d_in[9],  (const float*)d_in[7],  (const float*)d_in[8]  };
  const float* Wlin[3] = { (const float*)d_in[12], (const float*)d_in[10], (const float*)d_in[11] };
  const float* Wc[3]   = { (const float*)d_in[15], (const float*)d_in[13], (const float*)d_in[14] };
  const float* Wh[3]   = { (const float*)d_in[18], (const float*)d_in[16], (const float*)d_in[17] };
  const float* featsF[3] = { txt, aud, vis };
  float* outp = (float*)d_out;

  // -- bump allocator over d_ws --
  char* ws = (char*)d_ws;
  size_t off = 0;
  auto alloc = [&](size_t bytes) -> void* {
    void* p = ws + off;
    off += (bytes + 255) & ~(size_t)255;
    return p;
  };

  __bf16* txt_b = (__bf16*)alloc(BLD * 2);
  __bf16* aud_b = (__bf16*)alloc(BLD * 2);
  __bf16* vis_b = (__bf16*)alloc(BLD * 2);
  __bf16* txtT  = (__bf16*)alloc(BLD * 2);       // [B, D, L] transposed bf16
  __bf16* audT  = (__bf16*)alloc(BLD * 2);
  __bf16* visT  = (__bf16*)alloc(BLD * 2);
  __bf16* featsT[3] = { txtT, audT, visT };
  __bf16* Wi_b = (__bf16*)alloc(Dn * 2 * Dn * 2);
  __bf16* Wq_b = (__bf16*)alloc(Dn * 2 * Dn * 2);
  __bf16* Waff_b[3], *Wlin_b[3], *Wc_b[3], *Wh_b[3];
  for (int r = 0; r < 3; ++r) Waff_b[r] = (__bf16*)alloc(Ln * Ln * 2);
  for (int r = 0; r < 3; ++r) Wlin_b[r] = (__bf16*)alloc(Ln * Kn * 2);
  for (int r = 0; r < 3; ++r) Wc_b[r]   = (__bf16*)alloc(2 * Dn * Kn * 2);
  for (int r = 0; r < 3; ++r) Wh_b[r]   = (__bf16*)alloc(Kn * Ln * 2);
  float* p1 = (float*)alloc(BLD * 4);
  float* p2 = (float*)alloc(BLD * 4);
  float* zz = (float*)alloc((Bn * Dn + 4) * 4);     // [8192] per-(b,d) + [4] scalars
  float* scal = zz + Bn * Dn;
  __bf16* G_b  = (__bf16*)alloc(BLD * 2);
  __bf16* G_ag = (__bf16*)alloc(BL2D * 2);
  __bf16* G_lg = (__bf16*)alloc(BL2D * 2);
  __bf16* T_b  = (__bf16*)alloc(Bn * Dn * Ln * 2);
  __bf16* Ct_b = (__bf16*)alloc(Bn * Dn * Kn * 2);
  float*  Hlin = (float*)alloc(Bn * Dn * Kn * 4);
  __bf16* H_b  = (__bf16*)alloc(Bn * Dn * Kn * 2);

  // 1) zero norm accumulators (8196 floats)
  zero_k<<<(8196 + 255) / 256, 256, 0, stream>>>(zz, 8196);

  // 2) f32 -> bf16 conversions + feature transposes
  auto cvt = [&](const float* s, __bf16* d, long long n) {
    int blocks = (int)((n + 255) / 256);
    if (blocks > 4096) blocks = 4096;
    cvt_k<<<blocks, 256, 0, stream>>>(s, d, n);
  };
  cvt(txt, txt_b, BLD);
  cvt(aud, aud_b, BLD);
  cvt(vis, vis_b, BLD);
  cvt(Wi, Wi_b, Dn * 2 * Dn);
  cvt(Wq, Wq_b, Dn * 2 * Dn);
  for (int r = 0; r < 3; ++r) cvt(Waff[r], Waff_b[r], Ln * Ln);
  for (int r = 0; r < 3; ++r) cvt(Wlin[r], Wlin_b[r], Ln * Kn);
  for (int r = 0; r < 3; ++r) cvt(Wc[r],   Wc_b[r],   2 * Dn * Kn);
  for (int r = 0; r < 3; ++r) cvt(Wh[r],   Wh_b[r],   Kn * Ln);
  {
    dim3 tg(4, 32, 64), tb(32, 8);
    trans_k<<<tg, tb, 0, stream>>>(txt, txtT);
    trans_k<<<tg, tb, 0, stream>>>(aud, audT);
    trans_k<<<tg, tb, 0, stream>>>(vis, visT);
  }

  // 3) f1 = txt@Wi+bi, f2 = aud@Wq+bq, fused pairwise-channel-sum + sum(f^2)
  {
    dim3 grid(2 * Dn / BT, (unsigned)(Bn * Ln / BT), 1);
    gemm_bf16_k<2><<<grid, 128, 0, stream>>>(
        txt_b, Dn, 0,  Wi_b, 2 * Dn, 1, 0,
        p1, Dn, 1, 0,  bi,  nullptr, 0, 0, 0,  &scal[0],
        (int)Dn, 1.0f, 0);
    gemm_bf16_k<2><<<grid, 128, 0, stream>>>(
        aud_b, Dn, 0,  Wq_b, 2 * Dn, 1, 0,
        p2, Dn, 1, 0,  bq,  nullptr, 0, 0, 0,  &scal[1],
        (int)Dn, 1.0f, 0);
  }

  // 4) norm-fusion weights, z = w1*p1 + w2*p2 (aliased into p1), per-(b,d) ||z||^2
  scalars_k<<<1, 1, 0, stream>>>(scal);
  zkern<<<dim3(64, 8), 128, 0, stream>>>(p1, p2, scal, p1, zz);

  // 5) G = F.normalize(z, dim=L) -> bf16 ; concats
  norm_k<<<4096, 256, 0, stream>>>(p1, zz, G_b, BLD);
  concat_k<<<4096, 256, 0, stream>>>(txt_b, aud_b, G_b, G_ag, G_lg, BL2D);

  // 6) three branches (att reassociated: T = feats^T @ W_aff^T ; att = T @ G_src)
  for (int r = 0; r < 3; ++r) {
    const __bf16* fT = featsT[r];
    const float*  ff = featsF[r];
    const __bf16* Gsrc = (r == 0) ? G_lg : G_ag;
    float* outBranch = outp + (long long)r * BLD;

    // T[b,d,l'] = sum_l featsT[b,d,l] * Waff[l',l]   (M=128, N=1024, K=1024)
    gemm_bf16_k<1><<<dim3(16, 2, 64), 128, 0, stream>>>(
        fT, Ln, Dn * Ln,               // A = feats^T (pre-transposed, k-contiguous)
        Waff_b[r], 1, Ln, 0,           // B = Waff^T (k-contiguous)
        T_b, Ln, 1, Dn * Ln,
        nullptr, nullptr, 0, 0, 0, nullptr,
        (int)Ln, 1.0f, 0);

    // C = tanh( (T @ G_src) / sqrt(2D) )   (M=128, N=256, K=1024) -> bf16
    gemm_bf16_k<1><<<dim3(4, 2, 64), 128, 0, stream>>>(
        T_b, Ln, Dn * Ln,
        Gsrc, 2 * Dn, 1, Ln * 2 * Dn,
        Ct_b, Kn, 1, Dn * Kn,
        nullptr, nullptr, 0, 0, 0, nullptr,
        (int)Ln, 1.0f / 16.0f, 1);

    // Hlin[b,d,k] = sum_l featsT[b,d,l] * Wlin[l,k]  (M=128, N=256, K=1024) -> f32
    gemm_bf16_k<0><<<dim3(4, 2, 64), 128, 0, stream>>>(
        fT, Ln, Dn * Ln,
        Wlin_b[r], Kn, 1, 0,
        Hlin, Kn, 1, Dn * Kn,
        nullptr, nullptr, 0, 0, 0, nullptr,
        (int)Ln, 1.0f, 0);

    // H = relu(C @ Wc + Hlin)   (M=128, N=256, K=256) -> bf16
    gemm_bf16_k<1><<<dim3(4, 2, 64), 128, 0, stream>>>(
        Ct_b, Kn, Dn * Kn,
        Wc_b[r], Kn, 1, 0,
        H_b, Kn, 1, Dn * Kn,
        nullptr, Hlin, Kn, 1, Dn * Kn, nullptr,
        (int)Kn, 1.0f, 2);

    // out[b,l,d] = (H @ Wh)[d,l] + feats[b,l,d]   (M=128, N=1024, K=256),
    // transposed store -> LDS-transpose epilogue, coalesced along d
    gemm_bf16_k<0><<<dim3(16, 2, 64), 128, 0, stream>>>(
        H_b, Kn, Dn * Kn,
        Wh_b[r], Ln, 1, 0,
        outBranch, 1, Dn, Ln * Dn,
        nullptr, ff, 1, Dn, Ln * Dn, nullptr,
        (int)Kn, 1.0f, 0);
  }
}
